// BioTransformerBlock_51427938402927
// MI455X (gfx1250) — compile-verified, matrix operand order checked by reading, not run on hardware
//
#include <hip/hip_runtime.h>
#include <stdint.h>
#include <stddef.h>

// ---------------- problem constants ----------------
#define B_   2
#define C_   128
#define HH_  48
#define WW_  48
#define S_   2304   // 48*48
#define SC_  144    // 12*12
#define L_   16
#define NH_  8

typedef __bf16 bf16x16 __attribute__((ext_vector_type(16)));
typedef float f32x8 __attribute__((ext_vector_type(8)));

__device__ __forceinline__ uint16_t f2b(float f) {
  uint32_t u = __float_as_uint(f);
  u += 0x7fffu + ((u >> 16) & 1u);   // round-to-nearest-even
  return (uint16_t)(u >> 16);
}
__device__ __forceinline__ float b2f(uint16_t h) {
  return __uint_as_float(((uint32_t)h) << 16);
}

union FragBF { bf16x16 v; uint32_t u[8]; };

// ISA 7.12.2: 16-bit A (16x32) fragment: lane = {row = lane&15},
// lanes 0-15 hold K = 0..7,16..23 ; lanes 16-31 hold K = 8..15,24..31
// (pairs of consecutive K per dword). B (32x16) is the mirrored layout with
// row = column n, so one loader serves both when the tile is stored
// [row][k] row-major (i.e. B stored transposed: [n][k]).
__device__ __forceinline__ bf16x16 load_frag(const uint16_t* base, int rowStride) {
  int lane = threadIdx.x & 31;
  int r = lane & 15;
  int kb = (lane < 16) ? 0 : 8;
  FragBF f;
  const uint16_t* row = base + (size_t)r * rowStride;
#pragma unroll
  for (int j = 0; j < 8; ++j) {
    int k = kb + ((j < 4) ? (2 * j) : (16 + 2 * (j - 4)));
    f.u[j] = *(const uint32_t*)(row + k);
  }
  return f.v;
}

__device__ __forceinline__ f32x8 wmma_bf(bf16x16 a, bf16x16 b, f32x8 c) {
  return __builtin_amdgcn_wmma_f32_16x16x32_bf16(false, a, false, b, (short)0, c, false, false);
}
__device__ __forceinline__ f32x8 vzero8() {
  f32x8 z = {0.f, 0.f, 0.f, 0.f, 0.f, 0.f, 0.f, 0.f};
  return z;
}

// CDNA5 async DMA: copy 16B from a per-lane global address into a per-lane LDS
// byte offset; tracked by ASYNCcnt (ISA 15.18.3 op 98).
__device__ __forceinline__ void async_copy_b128(uint32_t ldsOff, const void* gptr) {
  asm volatile("global_load_async_to_lds_b128 %0, %1, off"
               :: "v"(ldsOff), "v"((unsigned long long)(size_t)gptr)
               : "memory");
}
__device__ __forceinline__ void wait_asynccnt0() {
  asm volatile("s_wait_asynccnt 0x0" ::: "memory");
}

// ---------------- f32 -> bf16 copy (weights) ----------------
__global__ void f2b_kernel(const float* __restrict__ in, uint16_t* __restrict__ out, int n) {
  int i = blockIdx.x * blockDim.x + threadIdx.x;
  if (i < n) out[i] = f2b(in[i]);
}

// ---------------- LayerNorm over channels -> bf16 ----------------
// one wave per spatial location, lane covers 4 channels
__global__ __launch_bounds__(256) void ln_kernel(const float* __restrict__ x,
                                                 const float* __restrict__ g,
                                                 const float* __restrict__ b,
                                                 uint16_t* __restrict__ out) {
  int wave = threadIdx.x >> 5, lane = threadIdx.x & 31;
  int loc = blockIdx.x * 8 + wave;          // [0, B_*S_)
  int bb = loc / S_, s = loc - bb * S_;
  float v[4];
  float sum = 0.f, sq = 0.f;
#pragma unroll
  for (int i = 0; i < 4; ++i) {
    int c = lane + 32 * i;
    float t = x[((size_t)bb * C_ + c) * S_ + s];
    v[i] = t; sum += t; sq += t * t;
  }
#pragma unroll
  for (int off = 16; off >= 1; off >>= 1) {
    sum += __shfl_xor(sum, off, 32);
    sq  += __shfl_xor(sq, off, 32);
  }
  float mean = sum * (1.0f / C_);
  float var = sq * (1.0f / C_) - mean * mean;
  float rinv = rsqrtf(var + 1e-5f);
#pragma unroll
  for (int i = 0; i < 4; ++i) {
    int c = lane + 32 * i;
    out[((size_t)bb * C_ + c) * S_ + s] = f2b((v[i] - mean) * rinv * g[c] + b[c]);
  }
}

// ---------------- q/k L2 normalize into padded (bh, sPad, 32) layout ----------------
__global__ void qknorm_kernel(const uint16_t* __restrict__ src, long long srcBatch,
                              int srcRow, int coff,
                              uint16_t* __restrict__ out, int seqLen, int sPad) {
  int idx = blockIdx.x * blockDim.x + threadIdx.x;
  if (idx >= B_ * NH_ * sPad) return;
  int s = idx % sPad;
  int bh = idx / sPad;
  int bb = bh >> 3, h = bh & 7;
  uint16_t* o = out + (size_t)idx * 32;
  if (s >= seqLen) {
#pragma unroll
    for (int i = 0; i < 32; ++i) o[i] = 0;
    return;
  }
  const uint16_t* p = src + (size_t)bb * srcBatch + (size_t)(coff + h * 16) * srcRow + s;
  float vv[16];
  float ss = 0.f;
#pragma unroll
  for (int d = 0; d < 16; ++d) {
    float t = b2f(p[(size_t)d * srcRow]);
    vv[d] = t; ss += t * t;
  }
  float inv = 1.0f / fmaxf(sqrtf(ss), 1e-12f);
#pragma unroll
  for (int d = 0; d < 16; ++d) o[d] = f2b(vv[d] * inv);
#pragma unroll
  for (int d = 16; d < 32; ++d) o[d] = 0;
}

// ---------------- 4x4 average pool (bf16 -> bf16) ----------------
__global__ void avgpool_kernel(const uint16_t* __restrict__ xn, uint16_t* __restrict__ xc) {
  int idx = blockIdx.x * blockDim.x + threadIdx.x;
  if (idx >= B_ * C_ * SC_) return;
  int ow = idx % 12;
  int t = idx / 12;
  int oh = t % 12;
  int bc = t / 12;
  const uint16_t* p = xn + (size_t)bc * S_ + (size_t)(oh * 4) * WW_ + ow * 4;
  float s = 0.f;
#pragma unroll
  for (int iy = 0; iy < 4; ++iy)
#pragma unroll
    for (int ix = 0; ix < 4; ++ix) s += b2f(p[iy * WW_ + ix]);
  xc[(size_t)bc * SC_ + oh * 12 + ow] = f2b(s * (1.0f / 16.0f));
}

// ---------------- bilinear 12x12 -> 48x48 (half-pixel), write into fuse_in ch 128.. ----------------
__global__ void bilinear_kernel(const float* __restrict__ g, uint16_t* __restrict__ fuse_in) {
  int idx = blockIdx.x * blockDim.x + threadIdx.x;
  if (idx >= B_ * C_ * S_) return;
  int x = idx % 48;
  int t = idx / 48;
  int y = t % 48;
  int bc = t / 48;
  int bb = bc / C_, c = bc - bb * C_;
  float sy = (y + 0.5f) * 0.25f - 0.5f;
  float sx = (x + 0.5f) * 0.25f - 0.5f;
  int y0 = (int)floorf(sy); float fy = sy - (float)y0;
  int x0 = (int)floorf(sx); float fx = sx - (float)x0;
  int y0c = min(max(y0, 0), 11), y1c = min(max(y0 + 1, 0), 11);
  int x0c = min(max(x0, 0), 11), x1c = min(max(x0 + 1, 0), 11);
  const float* p = g + (size_t)bc * SC_;
  float v00 = p[y0c * 12 + x0c], v01 = p[y0c * 12 + x1c];
  float v10 = p[y1c * 12 + x0c], v11 = p[y1c * 12 + x1c];
  float val = v00 * (1.f - fy) * (1.f - fx) + v01 * (1.f - fy) * fx +
              v10 * fy * (1.f - fx) + v11 * fy * fx;
  fuse_in[((size_t)bb * 2 * C_ + C_ + c) * S_ + (size_t)y * 48 + x] = f2b(val);
}

// ---------------- tiled GEMM: Y(MxN) = W(MxK) * X(KxN), bf16 in, f32 acc ----------------
// 64x64 tile per workgroup, 4 waves, K stepped by 32 (one WMMA per 16x16x32).
// A tile staged via global_load_async_to_lds_b128 (TDM-style async DMA path),
// B tile transposed manually (coalesced along N). 40-element LDS row stride
// keeps fragment dword loads 16B-aligned while rotating banks.
#define AST 40
#define BST 40

__global__ __launch_bounds__(128) void gemm_kernel(
    const uint16_t* __restrict__ W, const uint16_t* __restrict__ X,
    const float* __restrict__ bias,
    float* __restrict__ outF, const float* __restrict__ resid,
    uint16_t* __restrict__ outB,
    int M, int N, int K,
    long long xBatch, long long yBatchF, long long yBatchB, long long rBatch,
    int gelu) {
  __shared__ uint16_t As[64 * AST];   // A tile [m][k]
  __shared__ uint16_t Bs[64 * BST];   // B tile transposed: [n][k]
  int bb = blockIdx.z;
  int n0 = blockIdx.x * 64, m0 = blockIdx.y * 64;
  int tid = threadIdx.x, wave = tid >> 5, lane = tid & 31;
  const uint16_t* Xb = X + (size_t)bb * xBatch;
  const bool full = (n0 + 64 <= N);
  uint32_t asBase = (uint32_t)(size_t)&As[0];

  f32x8 acc[4];
#pragma unroll
  for (int j = 0; j < 4; ++j) acc[j] = vzero8();

  int bn = tid & 63;        // B staging: this thread's column
  int khalf = tid >> 6;     // 0: k 0..15, 1: k 16..31

  for (int k0 = 0; k0 < K; k0 += 32) {
    // ---- A tile: 64x32 bf16 = 256 x 16B chunks, async DMA to LDS ----
#pragma unroll
    for (int i = 0; i < 2; ++i) {
      int li = i * 128 + tid;
      int r = li >> 2, ch = li & 3;          // row, 8-element chunk
      uint32_t lds = asBase + (uint32_t)(r * AST + ch * 8) * 2;
      async_copy_b128(lds, &W[(size_t)(m0 + r) * K + k0 + ch * 8]);
    }
    // ---- B tile: transpose while async A copy is in flight ----
    if (full) {
#pragma unroll
      for (int k = 0; k < 16; ++k) {
        int kk = khalf * 16 + k;
        Bs[bn * BST + kk] = Xb[(size_t)(k0 + kk) * N + n0 + bn];
      }
    } else {
      int gn = n0 + bn;
#pragma unroll
      for (int k = 0; k < 16; ++k) {
        int kk = khalf * 16 + k;
        Bs[bn * BST + kk] = (gn < N) ? Xb[(size_t)(k0 + kk) * N + gn] : (uint16_t)0;
      }
    }
    wait_asynccnt0();
    __syncthreads();
    bf16x16 af = load_frag(&As[wave * 16 * AST], AST);
#pragma unroll
    for (int j = 0; j < 4; ++j) {
      bf16x16 bfm = load_frag(&Bs[j * 16 * BST], BST);
      acc[j] = wmma_bf(af, bfm, acc[j]);
    }
    __syncthreads();
  }

  // epilogue (C/D layout: lane = col n&15 (+16 -> rows +8), vgpr r = row)
  int ncol = lane & 15;
  int mbase = m0 + wave * 16 + ((lane < 16) ? 0 : 8);
#pragma unroll
  for (int j = 0; j < 4; ++j) {
#pragma unroll
    for (int r = 0; r < 8; ++r) {
      int m = mbase + r;
      int n = n0 + j * 16 + ncol;
      if (full || n < N) {
        float val = acc[j][r];
        if (bias) val += bias[m];
        if (gelu) val = 0.5f * val * (1.0f + erff(val * 0.70710678f));
        if (outF) {
          float o = val;
          if (resid) o += resid[(size_t)bb * rBatch + (size_t)m * N + n];
          outF[(size_t)bb * yBatchF + (size_t)m * N + n] = o;
        }
        if (outB) outB[(size_t)bb * yBatchB + (size_t)m * N + n] = f2b(val);
      }
    }
  }
}

// ---------------- flash cosine attention, one wave per 16-query block ----------------
// qn/kn: (bh, sPad, 32) bf16 (hd padded to 32); v: channel-major (d rows, key cols)
__global__ __launch_bounds__(32) void flash_kernel(
    const uint16_t* __restrict__ qn, const uint16_t* __restrict__ kn,
    const uint16_t* __restrict__ v, uint16_t* __restrict__ out,
    int seqq, int seqk, int skPad,
    long long vBatch, int vRow,
    long long oBatch, int oRow,
    float scale) {
  __shared__ uint16_t ptile[16 * 32];
  int bh = blockIdx.y;
  int bb = bh >> 3, h = bh & 7;
  int q0 = blockIdx.x * 16;
  int lane = threadIdx.x;

  bf16x16 qf = load_frag(qn + ((size_t)bh * seqq + q0) * 32, 32);
  const uint16_t* kbase = kn + (size_t)bh * skPad * 32;
  const uint16_t* vbase = v + (size_t)bb * vBatch + (size_t)(h * 16) * vRow;

  f32x8 acc = vzero8();
  f32x8 zero = vzero8();
  float mrow[8], lrow[8];
#pragma unroll
  for (int r = 0; r < 8; ++r) { mrow[r] = -3e38f; lrow[r] = 0.f; }

  for (int kb = 0; kb < seqk; kb += 32) {
    bf16x16 kf0 = load_frag(kbase + (size_t)kb * 32, 32);
    bf16x16 kf1 = load_frag(kbase + (size_t)(kb + 16) * 32, 32);
    f32x8 s0 = wmma_bf(qf, kf0, zero);
    f32x8 s1 = wmma_bf(qf, kf1, zero);

    int key0 = kb + (lane & 15), key1 = kb + 16 + (lane & 15);
    float a0[8], a1[8];
#pragma unroll
    for (int r = 0; r < 8; ++r) {
      a0[r] = (key0 < seqk) ? s0[r] * scale : -3e38f;
      a1[r] = (key1 < seqk) ? s1[r] * scale : -3e38f;
    }
#pragma unroll
    for (int r = 0; r < 8; ++r) {
      float mx = fmaxf(a0[r], a1[r]);
      mx = fmaxf(mx, __shfl_xor(mx, 1, 32));
      mx = fmaxf(mx, __shfl_xor(mx, 2, 32));
      mx = fmaxf(mx, __shfl_xor(mx, 4, 32));
      mx = fmaxf(mx, __shfl_xor(mx, 8, 32));
      float mnew = fmaxf(mrow[r], mx);
      float alpha = __expf(mrow[r] - mnew);
      float p0 = __expf(a0[r] - mnew);
      float p1 = __expf(a1[r] - mnew);
      a0[r] = p0; a1[r] = p1;
      float srow = p0 + p1;
      srow += __shfl_xor(srow, 1, 32);
      srow += __shfl_xor(srow, 2, 32);
      srow += __shfl_xor(srow, 4, 32);
      srow += __shfl_xor(srow, 8, 32);
      lrow[r] = lrow[r] * alpha + srow;
      mrow[r] = mnew;
      acc[r] = acc[r] * alpha;
    }
    // P (16x32) -> LDS in A-fragment friendly [m][k] layout
    int mr = (lane < 16) ? 0 : 8;
    int nc = lane & 15;
#pragma unroll
    for (int r = 0; r < 8; ++r) {
      ptile[(mr + r) * 32 + nc] = f2b(a0[r]);
      ptile[(mr + r) * 32 + 16 + nc] = f2b(a1[r]);
    }
    __syncthreads();
    bf16x16 pf = load_frag(&ptile[0], 32);
    bf16x16 vf = load_frag(vbase + kb, vRow);   // rows = d, K pairs = keys
    acc = wmma_bf(pf, vf, acc);
    __syncthreads();
  }

  int mr = (lane < 16) ? 0 : 8;
  int d = lane & 15;
  uint16_t* ob = out + (size_t)bb * oBatch + (size_t)(h * 16 + d) * oRow;
#pragma unroll
  for (int r = 0; r < 8; ++r) {
    int q = q0 + mr + r;
    if (q < seqq) ob[q] = f2b(acc[r] / lrow[r]);
  }
}

// ---------------- host orchestration ----------------
static inline void launch_gemm(hipStream_t st, const uint16_t* Wb, const uint16_t* X,
                               const float* bias, float* outF, const float* resid,
                               uint16_t* outB, int M, int N, int K,
                               long long xB, long long yBF, long long yBB, long long rB,
                               int gelu, int batches) {
  dim3 g((N + 63) / 64, M / 64, batches);
  gemm_kernel<<<g, 128, 0, st>>>(Wb, X, bias, outF, resid, outB,
                                 M, N, K, xB, yBF, yBB, rB, gelu);
}

extern "C" void kernel_launch(void* const* d_in, const int* in_sizes, int n_in,
                              void* d_out, int out_size, void* d_ws, size_t ws_size,
                              hipStream_t stream) {
  (void)in_sizes; (void)n_in; (void)out_size; (void)ws_size;
  const float* x0 = (const float*)d_in[0];

  // workspace cursor (256B aligned)
  size_t off = 0;
  char* wsb = (char*)d_ws;
  auto alloc = [&](size_t bytes) -> char* {
    char* p = wsb + off;
    off += (bytes + 255) & ~(size_t)255;
    return p;
  };

  // bf16 weight copies
  struct WCvt { int idx; int n; };
  const WCvt wc[14] = {
      {3, 384 * 128}, {4, 128 * 128}, {6, 384 * 128}, {7, 128 * 128},
      {9, 128 * 256}, {11, 128 * 128}, {16, 128 * 128}, {17, 256 * 128},
      {18, 128 * 128}, {19, 256 * 128}, {20, 128 * 128}, {24, 512 * 128},
      {26, 128 * 512}, {15, 128 * 16}};
  uint16_t* wb[14];
  for (int i = 0; i < 14; ++i) wb[i] = (uint16_t*)alloc((size_t)wc[i].n * 2);

  uint16_t* xn       = (uint16_t*)alloc((size_t)B_ * C_ * S_ * 2);
  uint16_t* qkv_loc  = (uint16_t*)alloc((size_t)B_ * 3 * C_ * S_ * 2);
  uint16_t* qn_loc   = (uint16_t*)alloc((size_t)B_ * NH_ * S_ * 32 * 2);
  uint16_t* kn_loc   = (uint16_t*)alloc((size_t)B_ * NH_ * S_ * 32 * 2);
  uint16_t* attnloc  = (uint16_t*)alloc((size_t)B_ * C_ * S_ * 2);
  uint16_t* fuse_in  = (uint16_t*)alloc((size_t)B_ * 2 * C_ * S_ * 2);
  uint16_t* xc       = (uint16_t*)alloc((size_t)B_ * C_ * SC_ * 2);
  uint16_t* qkv_g    = (uint16_t*)alloc((size_t)B_ * 3 * C_ * SC_ * 2);
  uint16_t* qn_g     = (uint16_t*)alloc((size_t)B_ * NH_ * SC_ * 32 * 2);
  uint16_t* kn_g     = (uint16_t*)alloc((size_t)B_ * NH_ * 160 * 32 * 2);
  uint16_t* attng    = (uint16_t*)alloc((size_t)B_ * C_ * SC_ * 2);
  float*    gglob    = (float*)alloc((size_t)B_ * C_ * SC_ * 4);
  uint16_t* fuse_mid = (uint16_t*)alloc((size_t)B_ * C_ * S_ * 2);
  float*    x1       = (float*)alloc((size_t)B_ * C_ * S_ * 4);
  float*    x2       = (float*)alloc((size_t)B_ * C_ * S_ * 4);
  uint16_t* qlb      = (uint16_t*)alloc((size_t)C_ * L_ * 2);
  uint16_t* qn_lat   = (uint16_t*)alloc((size_t)B_ * NH_ * L_ * 32 * 2);
  uint16_t* kvx      = (uint16_t*)alloc((size_t)B_ * 2 * C_ * S_ * 2);
  uint16_t* kn_x     = (uint16_t*)alloc((size_t)B_ * NH_ * S_ * 32 * 2);
  uint16_t* latread  = (uint16_t*)alloc((size_t)B_ * C_ * L_ * 2);
  uint16_t* kvlat    = (uint16_t*)alloc((size_t)B_ * 2 * C_ * L_ * 2);
  uint16_t* qxb      = (uint16_t*)alloc((size_t)B_ * C_ * S_ * 2);
  uint16_t* qn_x     = (uint16_t*)alloc((size_t)B_ * NH_ * S_ * 32 * 2);
  uint16_t* kn_l     = (uint16_t*)alloc((size_t)B_ * NH_ * 32 * 32 * 2);
  uint16_t* xw       = (uint16_t*)alloc((size_t)B_ * C_ * S_ * 2);
  uint16_t* ffnmid   = (uint16_t*)alloc((size_t)B_ * 4 * C_ * S_ * 2);

  // convert weights to bf16
  for (int i = 0; i < 14; ++i) {
    int n = wc[i].n;
    f2b_kernel<<<(n + 255) / 256, 256, 0, stream>>>((const float*)d_in[wc[i].idx], wb[i], n);
  }
  uint16_t* w_locqkv = wb[0];  uint16_t* w_locproj = wb[1];
  uint16_t* w_gqkv = wb[2];    uint16_t* w_gproj = wb[3];
  uint16_t* w_f1 = wb[4];      uint16_t* w_f2 = wb[5];
  uint16_t* w_qlat = wb[6];    uint16_t* w_kvx = wb[7];
  uint16_t* w_qx = wb[8];      uint16_t* w_kvlat = wb[9];
  uint16_t* w_latproj = wb[10]; uint16_t* w_ffn1 = wb[11];
  uint16_t* w_ffn2 = wb[12];   uint16_t* latbf = wb[13];

  const long long CS = (long long)C_ * S_;       // 294912
  const long long CS3 = 3 * CS, CS2 = 2 * CS, CS4 = 4 * CS;
  const long long CSc = (long long)C_ * SC_;     // 18432

  // === BioAttentionFusion ===
  ln_kernel<<<(B_ * S_) / 8, 256, 0, stream>>>(x0, (const float*)d_in[1], (const float*)d_in[2], xn);
  launch_gemm(stream, w_locqkv, xn, nullptr, nullptr, nullptr, qkv_loc,
              384, S_, 128, CS, 0, CS3, 0, 0, B_);
  {
    int tot = B_ * NH_ * S_;
    qknorm_kernel<<<(tot + 255) / 256, 256, 0, stream>>>(qkv_loc, CS3, S_, 0, qn_loc, S_, S_);
    qknorm_kernel<<<(tot + 255) / 256, 256, 0, stream>>>(qkv_loc, CS3, S_, C_, kn_loc, S_, S_);
  }
  flash_kernel<<<dim3(S_ / 16, B_ * NH_), 32, 0, stream>>>(
      qn_loc, kn_loc, qkv_loc + (size_t)2 * C_ * S_, attnloc,
      S_, S_, S_, CS3, S_, CS, S_, 0.25f);
  launch_gemm(stream, w_locproj, attnloc, (const float*)d_in[5], nullptr, nullptr, fuse_in,
              128, S_, 128, CS, 0, CS2, 0, 0, B_);
  avgpool_kernel<<<(B_ * C_ * SC_ + 255) / 256, 256, 0, stream>>>(xn, xc);
  launch_gemm(stream, w_gqkv, xc, nullptr, nullptr, nullptr, qkv_g,
              384, SC_, 128, CSc, 0, 3 * CSc, 0, 0, B_);
  {
    int totq = B_ * NH_ * SC_;
    int totk = B_ * NH_ * 160;
    qknorm_kernel<<<(totq + 255) / 256, 256, 0, stream>>>(qkv_g, 3 * CSc, SC_, 0, qn_g, SC_, SC_);
    qknorm_kernel<<<(totk + 255) / 256, 256, 0, stream>>>(qkv_g, 3 * CSc, SC_, C_, kn_g, SC_, 160);
  }
  flash_kernel<<<dim3(SC_ / 16, B_ * NH_), 32, 0, stream>>>(
      qn_g, kn_g, qkv_g + (size_t)2 * C_ * SC_, attng,
      SC_, SC_, 160, 3 * CSc, SC_, CSc, SC_, 0.25f);
  launch_gemm(stream, w_gproj, attng, (const float*)d_in[8], gglob, nullptr, nullptr,
              128, SC_, 128, CSc, CSc, 0, 0, 0, B_);
  bilinear_kernel<<<(B_ * C_ * S_ + 255) / 256, 256, 0, stream>>>(gglob, fuse_in);
  launch_gemm(stream, w_f1, fuse_in, (const float*)d_in[10], nullptr, nullptr, fuse_mid,
              128, S_, 256, CS2, 0, CS, 0, 1, B_);
  launch_gemm(stream, w_f2, fuse_mid, (const float*)d_in[12], x1, x0, nullptr,
              128, S_, 128, CS, CS, 0, CS, 0, B_);

  // === LatentMixer ===
  ln_kernel<<<(B_ * S_) / 8, 256, 0, stream>>>(x1, (const float*)d_in[13], (const float*)d_in[14], xn);
  launch_gemm(stream, w_qlat, latbf, nullptr, nullptr, nullptr, qlb,
              128, L_, 128, 0, 0, 0, 0, 0, 1);
  {
    int tot = B_ * NH_ * L_;
    qknorm_kernel<<<(tot + 255) / 256, 256, 0, stream>>>(qlb, 0, L_, 0, qn_lat, L_, L_);
  }
  launch_gemm(stream, w_kvx, xn, nullptr, nullptr, nullptr, kvx,
              256, S_, 128, CS, 0, CS2, 0, 0, B_);
  {
    int tot = B_ * NH_ * S_;
    qknorm_kernel<<<(tot + 255) / 256, 256, 0, stream>>>(kvx, CS2, S_, 0, kn_x, S_, S_);
  }
  flash_kernel<<<dim3(L_ / 16, B_ * NH_), 32, 0, stream>>>(
      qn_lat, kn_x, kvx + (size_t)C_ * S_, latread,
      L_, S_, S_, CS2, S_, (long long)C_ * L_, L_, 0.25f);
  launch_gemm(stream, w_kvlat, latread, nullptr, nullptr, nullptr, kvlat,
              256, L_, 128, (long long)C_ * L_, 0, (long long)2 * C_ * L_, 0, 0, B_);
  launch_gemm(stream, w_qx, xn, nullptr, nullptr, nullptr, qxb,
              128, S_, 128, CS, 0, CS, 0, 0, B_);
  {
    int tot = B_ * NH_ * S_;
    qknorm_kernel<<<(tot + 255) / 256, 256, 0, stream>>>(qxb, CS, S_, 0, qn_x, S_, S_);
    int totk = B_ * NH_ * 32;
    qknorm_kernel<<<(totk + 255) / 256, 256, 0, stream>>>(kvlat, (long long)2 * C_ * L_, L_, 0, kn_l, L_, 32);
  }
  flash_kernel<<<dim3(S_ / 16, B_ * NH_), 32, 0, stream>>>(
      qn_x, kn_l, kvlat + (size_t)C_ * L_, xw,
      S_, L_, 32, (long long)2 * C_ * L_, L_, CS, S_, 0.25f);
  launch_gemm(stream, w_latproj, xw, (const float*)d_in[21], x2, x1, nullptr,
              128, S_, 128, CS, CS, 0, CS, 0, B_);

  // === FFN ===
  ln_kernel<<<(B_ * S_) / 8, 256, 0, stream>>>(x2, (const float*)d_in[22], (const float*)d_in[23], xn);
  launch_gemm(stream, w_ffn1, xn, (const float*)d_in[25], nullptr, nullptr, ffnmid,
              512, S_, 128, CS, 0, CS4, 0, 1, B_);
  launch_gemm(stream, w_ffn2, ffnmid, (const float*)d_in[27], (float*)d_out, x2, nullptr,
              128, S_, 512, CS4, CS, 0, CS, 0, B_);
}
